// Attention_68118181315226
// MI455X (gfx1250) — matrix operand order repaired and checked by reading, not compiled, not run on hardware
//
#include <hip/hip_runtime.h>
#include <hip/hip_bf16.h>
#include <math.h>

#define B_ 64
#define S_ 2048
#define D_ 512
#define H_ 512
#define APAD 8   // pad (in bf16 elems) -> 520*2B = 260 dwords row stride, conflict-free

typedef __attribute__((ext_vector_type(16))) __bf16 v16bf;
typedef __attribute__((ext_vector_type(8)))  float  v8f;

__device__ __forceinline__ float fast_tanh(float x) {
#if __has_builtin(__builtin_amdgcn_tanhf)
  return __builtin_amdgcn_tanhf(x);          // v_tanh_f32 (gfx1250 trans op)
#elif __has_builtin(__builtin_amdgcn_tanh_f32)
  return __builtin_amdgcn_tanh_f32(x);
#else
  return tanhf(x);
#endif
}

// ---------------------------------------------------------------------------
// 0) Pre-pack W_ctx into WMMA-B-fragment-ready bf16 hi/lo layout:
//    Bpack[nt][ks][lane][0..15]=hi, [16..31]=lo with
//    (r=lane&15, half=lane>>4): elem e -> B[k=ks*32+half*16+e][n=nt*16+r]
//                                      = W_ctx[nt*16+r][ks*32+half*16+e]
//    Each lane's fragment = 64 contiguous bytes; fully coalesced b128 loads.
// ---------------------------------------------------------------------------
__global__ __launch_bounds__(256) void pack_wctx_kernel(const float* __restrict__ W_ctx,
                                                        __bf16* __restrict__ Bpack) {
  int tid  = blockIdx.x * 256 + threadIdx.x;  // 32 nt * 16 ks * 32 lane * 16 e
  int e    = tid & 15;
  int lane = (tid >> 4) & 31;
  int ks   = (tid >> 9) & 15;
  int nt   = tid >> 13;
  int r    = lane & 15;
  int half = lane >> 4;
  float  f  = W_ctx[(size_t)(nt * 16 + r) * D_ + ks * 32 + half * 16 + e];
  __bf16 hi = (__bf16)f;
  size_t base = ((size_t)((nt * 16 + ks) * 32 + lane)) * 32;
  Bpack[base + e]      = hi;
  Bpack[base + 16 + e] = (__bf16)(f - (float)hi);
}

// ---------------------------------------------------------------------------
// 1) inp2[b,h] = x[b,:] . W_in[h,:] + b_in[h] + b_ctx[h]   (b_ctx folded in)
// ---------------------------------------------------------------------------
__global__ __launch_bounds__(256) void inp_kernel(const float* __restrict__ x,
                                                  const float* __restrict__ W,
                                                  const float* __restrict__ bias,
                                                  const float* __restrict__ b_ctx,
                                                  float* __restrict__ out) {
  int idx = blockIdx.x * 256 + threadIdx.x;   // b*H + h
  int b = idx >> 9;
  int h = idx & (H_ - 1);
  float acc = bias[h] + b_ctx[h];
  const float* xr = x + b * D_;
  const float* wr = W + h * D_;
#pragma unroll 4
  for (int d = 0; d < D_; ++d) acc += xr[d] * wr[d];
  out[idx] = acc;
}

// ---------------------------------------------------------------------------
// 2) Score kernel: att[b,s] = sum_h V[h]*tanh(inp2[b,h] + ctx[b,h,s])
//    ctx tile via v_wmma_f32_16x16x32_bf16 with hi/lo split (~fp32 accuracy).
//    Block: 256 thr (8 waves) = one batch x 16 seq rows; waves split H into
//    8*4 = 32 n-tiles of 16.
// ---------------------------------------------------------------------------
__global__ __launch_bounds__(256) void score_kernel(const float* __restrict__ context,
                                                    const __bf16* __restrict__ Bpack,
                                                    const float* __restrict__ V,
                                                    const float* __restrict__ inp2,
                                                    const unsigned char* __restrict__ mask,
                                                    float* __restrict__ att) {
  const int sb = blockIdx.x * 16;   // seq-tile base
  const int b  = blockIdx.y;

  __shared__ __bf16 As_hi[16][D_ + APAD];
  __shared__ __bf16 As_lo[16][D_ + APAD];
  __shared__ float  att_lds[16];

  const int t = threadIdx.x;

  // Stage 16 x 512 context rows as bf16 hi/lo (f32 ~= hi + lo).
  const float* cbase = context + ((size_t)b * S_ + sb) * D_;
  for (int i = t; i < 16 * D_; i += 256) {
    int row = i >> 9;
    int col = i & (D_ - 1);
    float f  = cbase[row * D_ + col];
    __bf16 hi = (__bf16)f;
    As_hi[row][col] = hi;
    As_lo[row][col] = (__bf16)(f - (float)hi);
  }
  if (t < 16) att_lds[t] = 0.0f;
  __syncthreads();

  const int wave = t >> 5;
  const int lane = t & 31;
  const int r    = lane & 15;    // 0..15
  const int half = lane >> 4;    // 0..1

  // attP[j] accumulates score contribution for seq row (j + 8*half)
  float attP[8];
#pragma unroll
  for (int j = 0; j < 8; ++j) attP[j] = 0.0f;

  for (int nt = 0; nt < 4; ++nt) {
    const int ntile = wave * 4 + nt;     // 0..31
    const int n     = ntile * 16 + r;    // this lane's h column

    v8f acc = {};
    for (int ks = 0; ks < D_ / 32; ++ks) {
      const int k0 = ks * 32;
      // --- A fragment (16x32 bf16): lane(r,half): elems 0..7 -> K=k0+16*half+e,
      //     elems 8..15 -> K=k0+16*half+8+(e-8)  (contiguous order, matches Bpack)
      v16bf a_hi, a_lo;
      const int ab1 = k0 + half * 16;
      const int ab2 = k0 + half * 16 + 8;
#pragma unroll
      for (int e = 0; e < 8; ++e) {
        a_hi[e]     = As_hi[r][ab1 + e];
        a_hi[e + 8] = As_hi[r][ab2 + e];
        a_lo[e]     = As_lo[r][ab1 + e];
        a_lo[e + 8] = As_lo[r][ab2 + e];
      }
      // --- B fragment: pre-packed, contiguous 64B per lane (L2-resident)
      const __bf16* bp = Bpack + ((size_t)((ntile * 16 + ks) * 32 + lane)) * 32;
      v16bf b_hi = *(const v16bf*)(bp);
      v16bf b_lo = *(const v16bf*)(bp + 16);

      // 3-term bf16 emulation of f32 GEMM: hi*hi + hi*lo + lo*hi
      acc = __builtin_amdgcn_wmma_f32_16x16x32_bf16(false, a_hi, false, b_hi,
                                                    (short)0, acc, false, false);
      acc = __builtin_amdgcn_wmma_f32_16x16x32_bf16(false, a_hi, false, b_lo,
                                                    (short)0, acc, false, false);
      acc = __builtin_amdgcn_wmma_f32_16x16x32_bf16(false, a_lo, false, b_hi,
                                                    (short)0, acc, false, false);
    }

    // acc[j] = ctx[b, h=n, s = sb + j + 8*half]
    const float inpv = inp2[b * H_ + n];
    const float vv   = V[n];
#pragma unroll
    for (int j = 0; j < 8; ++j)
      attP[j] += vv * fast_tanh(acc[j] + inpv);
  }

  // Reduce over the 16 h-columns held by this half (wave32 butterfly),
  // then fold into the per-row LDS accumulator.
#pragma unroll
  for (int j = 0; j < 8; ++j) {
    float v = attP[j];
    for (int off = 8; off > 0; off >>= 1) v += __shfl_xor(v, off, 32);
    if (r == 0) atomicAdd(&att_lds[j + half * 8], v);
  }
  __syncthreads();

  if (t < 16) {
    int s = sb + t;
    bool m = mask[b * S_ + s] != 0;
    att[b * S_ + s] = m ? -INFINITY : att_lds[t];
  }
}

// ---------------------------------------------------------------------------
// 3) Row softmax over S per batch -> alpha
// ---------------------------------------------------------------------------
__global__ __launch_bounds__(256) void softmax_kernel(const float* __restrict__ att,
                                                      float* __restrict__ alpha) {
  const int b = blockIdx.x;
  const int t = threadIdx.x;
  __shared__ float red[256];

  float m = -INFINITY;
  for (int s = t; s < S_; s += 256) m = fmaxf(m, att[b * S_ + s]);
  red[t] = m;
  __syncthreads();
  for (int o = 128; o > 0; o >>= 1) {
    if (t < o) red[t] = fmaxf(red[t], red[t + o]);
    __syncthreads();
  }
  const float mx = red[0];
  __syncthreads();

  float sum = 0.0f;
  for (int s = t; s < S_; s += 256) sum += __expf(att[b * S_ + s] - mx);
  red[t] = sum;
  __syncthreads();
  for (int o = 128; o > 0; o >>= 1) {
    if (t < o) red[t] += red[t + o];
    __syncthreads();
  }
  const float inv = 1.0f / red[0];
  for (int s = t; s < S_; s += 256)
    alpha[b * S_ + s] = __expf(att[b * S_ + s] - mx) * inv;
}

// ---------------------------------------------------------------------------
// 4) y[b,d] = sum_s alpha[b,s] * context[b,s,d]   (streaming, coalesced in d)
// ---------------------------------------------------------------------------
__global__ __launch_bounds__(256) void zero_kernel(float* __restrict__ p, int n) {
  int i = blockIdx.x * 256 + threadIdx.x;
  if (i < n) p[i] = 0.0f;
}

__global__ __launch_bounds__(512) void wsum_kernel(const float* __restrict__ context,
                                                   const float* __restrict__ alpha,
                                                   float* __restrict__ y) {
  const int b  = blockIdx.x;
  const int sc = blockIdx.y;        // 8 chunks of 256 seq positions
  const int d  = threadIdx.x;       // 0..511
  const int s0 = sc * 256;
  float acc = 0.0f;
  const float* cb = context + ((size_t)b * S_ + s0) * D_ + d;
  const float* ab = alpha + b * S_ + s0;
#pragma unroll 4
  for (int s = 0; s < 256; ++s) acc += ab[s] * cb[(size_t)s * D_];
  atomicAdd(&y[b * D_ + d], acc);
}

// ---------------------------------------------------------------------------
// 5) hidden[b,h] = y[b,:] . W_ctx[h,:] + b_ctx[h]
// ---------------------------------------------------------------------------
__global__ __launch_bounds__(256) void hidden_kernel(const float* __restrict__ y,
                                                     const float* __restrict__ W_ctx,
                                                     const float* __restrict__ b_ctx,
                                                     float* __restrict__ hidden) {
  int idx = blockIdx.x * 256 + threadIdx.x;   // b*H + h
  int b = idx >> 9;
  int h = idx & (H_ - 1);
  float acc = b_ctx[h];
  const float* yr = y + b * D_;
  const float* wr = W_ctx + (size_t)h * D_;
#pragma unroll 4
  for (int d = 0; d < D_; ++d) acc += yr[d] * wr[d];
  hidden[idx] = acc;
}

// ---------------------------------------------------------------------------
extern "C" void kernel_launch(void* const* d_in, const int* in_sizes, int n_in,
                              void* d_out, int out_size, void* d_ws, size_t ws_size,
                              hipStream_t stream) {
  const float*         x       = (const float*)d_in[0];
  const float*         context = (const float*)d_in[1];
  const unsigned char* mask    = (const unsigned char*)d_in[2];  // jax bool = 1 byte
  const float*         W_in    = (const float*)d_in[3];
  const float*         b_in    = (const float*)d_in[4];
  const float*         W_ctx   = (const float*)d_in[5];
  const float*         b_ctx   = (const float*)d_in[6];
  const float*         V       = (const float*)d_in[7];

  float* hidden = (float*)d_out;            // [B,H]
  float* alpha  = (float*)d_out + B_ * H_;  // [B,S]

  float*  inp   = (float*)d_ws;             // [B,H]  (= x@W_in^T + b_in + b_ctx)
  float*  att   = inp + B_ * H_;            // [B,S]  raw masked scores
  float*  y     = att + B_ * S_;            // [B,D]
  __bf16* Bpack = (__bf16*)(y + B_ * D_);   // 32*16*32*32 bf16 = 1 MB

  pack_wctx_kernel<<<dim3((32 * 16 * 32 * 16) / 256), 256, 0, stream>>>(W_ctx, Bpack);
  inp_kernel<<<dim3((B_ * H_) / 256), 256, 0, stream>>>(x, W_in, b_in, b_ctx, inp);
  score_kernel<<<dim3(S_ / 16, B_), 256, 0, stream>>>(context, Bpack, V, inp, mask, att);
  softmax_kernel<<<dim3(B_), 256, 0, stream>>>(att, alpha);
  zero_kernel<<<dim3((B_ * D_) / 256), 256, 0, stream>>>(y, B_ * D_);
  wsum_kernel<<<dim3(B_, 8), 512, 0, stream>>>(context, alpha, y);
  hidden_kernel<<<dim3((B_ * H_) / 256), 256, 0, stream>>>(y, W_ctx, b_ctx, hidden);
}
